// Injector_51393578664114
// MI455X (gfx1250) — matrix operand order repaired and checked by reading, not compiled, not run on hardware
//
#include <hip/hip_runtime.h>

typedef __attribute__((ext_vector_type(16))) _Float16 v16h;
typedef __attribute__((ext_vector_type(8)))  _Float16 v8h;
typedef __attribute__((ext_vector_type(4)))  _Float16 v4h;
typedef __attribute__((ext_vector_type(8)))  float    v8f;

#define HH  6
#define CC  768
#define DD  128
#define LQV 4096
#define NBV 4
#define LIN 21504

#define LDSP       776                      // 768 + 8 halves pad (bank spread)
#define GEMM_LDSB  (64 * LDSP * 2)          // 99,328 bytes dynamic LDS

// ------------------------ LayerNorm -> f16 ------------------------
__global__ void ln_f16_kernel(const float* __restrict__ x,
                              const float* __restrict__ w,
                              const float* __restrict__ b,
                              _Float16* __restrict__ out)
{
    __shared__ float s1[256];
    __shared__ float s2[256];
    const int row = blockIdx.x;
    const int t   = threadIdx.x;
    const float* xr = x + (size_t)row * CC;
    float v0 = xr[t], v1 = xr[t + 256], v2 = xr[t + 512];
    s1[t] = v0 + v1 + v2;
    s2[t] = v0 * v0 + v1 * v1 + v2 * v2;
    __syncthreads();
    for (int o = 128; o > 0; o >>= 1) {
        if (t < o) { s1[t] += s1[t + o]; s2[t] += s2[t + o]; }
        __syncthreads();
    }
    const float mean = s1[0] * (1.0f / CC);
    const float var  = s2[0] * (1.0f / CC) - mean * mean;
    const float rs   = rsqrtf(var + 1e-6f);
    _Float16* orow = out + (size_t)row * CC;
    float vv[3] = { v0, v1, v2 };
#pragma unroll
    for (int k = 0; k < 3; k++) {
        const int c = t + k * 256;
        orow[c] = (_Float16)((vv[k] - mean) * rs * w[c] + b[c]);
    }
}

// ------------------------ f32 -> f16 copy ------------------------
__global__ void cvt_f16_kernel(const float* __restrict__ s, _Float16* __restrict__ d, int n)
{
    const int i = blockIdx.x * blockDim.x + threadIdx.x;
    if (i < n) d[i] = (_Float16)s[i];
}

// ---- combined sampling-offset + attention-logit weights, padded to 256 rows ----
__global__ void build_wc_kernel(const float* __restrict__ so_w,
                                const float* __restrict__ aw_w,
                                const float* __restrict__ so_b,
                                const float* __restrict__ aw_b,
                                _Float16* __restrict__ wc,
                                float* __restrict__ cb)
{
    const int i = blockIdx.x * blockDim.x + threadIdx.x;
    if (i < 256 * CC) {
        const int r = i / CC, c = i - r * CC;
        float v = 0.f;
        if (r < 144)      v = so_w[(size_t)r * CC + c];
        else if (r < 216) v = aw_w[(size_t)(r - 144) * CC + c];
        wc[i] = (_Float16)v;
    }
    if (i < 256) {
        float bb = 0.f;
        if (i < 144)      bb = so_b[i];
        else if (i < 216) bb = aw_b[i - 144];
        cb[i] = bb;
    }
}

// ------------------------ WMMA GEMM: C = A(MxK) * Bw(NxK)^T ------------------------
// 128 threads = 4 waves (2M x 2N); block owns 64 rows of A, staged ONCE into LDS
// via async global->LDS, then loops over all 128-wide n-tiles (A read from HBM once).
// Wave tile = 32x64 -> 8 wmma per 32-deep K step; B ping-pong buffered in registers.
// __launch_bounds__(128,1): single-workgroup-per-EU target -> big VGPR budget, no spills.
// MODE 0: f16 store (acc+bias) | MODE 1: f32 store (acc+bias)
// MODE 2: f32 store resid + gamma*(acc+bias)
template <int MODE>
__global__ __launch_bounds__(128, 1)
void gemm_wmma_kernel(const _Float16* __restrict__ A,
                      const _Float16* __restrict__ Bw,
                      const float* __restrict__ bias,
                      void* __restrict__ Cout,
                      int M, int N, int K,
                      const float* __restrict__ resid,
                      const float* __restrict__ gamma)
{
    extern __shared__ _Float16 smem[];           // 64 x LDSP halves
    const int lane = threadIdx.x & 31;
    const int wv   = threadIdx.x >> 5;           // 0..3
    const int wm   = wv >> 1;                    // 0..1 : 32-row band
    const int wn   = wv & 1;                     // 0..1 : 64-col band
    const int l15  = lane & 15;
    const int hi   = lane >> 4;
    const int ka_off = hi << 4;                  // A frag: lanes 16-31 carry K+16
    const int kb_off = hi << 4;                  // B frag: lanes 16-31 carry K+16
    const int gm0  = blockIdx.x * 64;

    // ---- stage A panel (64 x K f16) into LDS with async global->LDS copies ----
    for (int idx = threadIdx.x; idx < 64 * (CC / 8); idx += 128) {
        const int row = idx / (CC / 8);
        const int c8  = idx - row * (CC / 8);
        const unsigned ldsoff =
            (unsigned)(((size_t)row * LDSP + (size_t)c8 * 8) * sizeof(_Float16));
        const unsigned long long ga =
            (unsigned long long)(uintptr_t)(A + (size_t)(gm0 + row) * K + (size_t)c8 * 8);
        asm volatile("global_load_async_to_lds_b128 %0, %1, off"
                     :: "v"(ldsoff), "v"(ga) : "memory");
    }
    asm volatile("s_wait_asynccnt 0" ::: "memory");
    __syncthreads();

    const int nTiles = N >> 7;
    for (int nt = 0; nt < nTiles; nt++) {
        const int n0 = nt * 128 + wn * 64;
        const _Float16* brow = Bw + (size_t)(n0 + l15) * K;

        v8f acc[2][4] = {};
        v16h b0[4], b1[4];

        auto loadB = [&](int k0, v16h (&dst)[4]) {
#pragma unroll
            for (int j = 0; j < 4; j++) {
                const _Float16* bp = brow + (size_t)(j << 4) * K + k0 + kb_off;
                v8h blo = *(const v8h*)(bp);
                v8h bhi = *(const v8h*)(bp + 8);
#pragma unroll
                for (int i = 0; i < 8; i++) { dst[j][i] = blo[i]; dst[j][i + 8] = bhi[i]; }
            }
        };
        auto compute = [&](int k0, v16h (&bb)[4]) {
            v16h a[2];
#pragma unroll
            for (int i = 0; i < 2; i++) {
                const _Float16* ap =
                    smem + (size_t)(wm * 32 + i * 16 + l15) * LDSP + k0 + ka_off;
                v8h alo = *(const v8h*)(ap);
                v8h ahi = *(const v8h*)(ap + 8);
#pragma unroll
                for (int q = 0; q < 8; q++) { a[i][q] = alo[q]; a[i][q + 8] = ahi[q]; }
            }
#pragma unroll
            for (int i = 0; i < 2; i++)
#pragma unroll
                for (int j = 0; j < 4; j++)
                    acc[i][j] = __builtin_amdgcn_wmma_f32_16x16x32_f16(
                        false, a[i], false, bb[j], (short)0, acc[i][j], false, false);
        };

        // K ping-pong: load b1 for k+32 while computing with b0, and vice versa
        loadB(0, b0);
        for (int k0 = 0; k0 < K; k0 += 64) {
            if (k0 + 32 < K) loadB(k0 + 32, b1);
            compute(k0, b0);
            if (k0 + 64 < K) loadB(k0 + 64, b0);
            if (k0 + 32 < K) compute(k0 + 32, b1);
        }

        // ---- epilogue ----
#pragma unroll
        for (int i = 0; i < 2; i++) {
            const int mbase = gm0 + wm * 32 + i * 16 + hi * 8;
#pragma unroll
            for (int j = 0; j < 4; j++) {
                const int n = n0 + (j << 4) + l15;
                const float bv = bias[n];
#pragma unroll
                for (int r = 0; r < 8; r++) {
                    const size_t o = (size_t)(mbase + r) * N + n;
                    const float v = acc[i][j][r] + bv;
                    if (MODE == 0)      ((_Float16*)Cout)[o] = (_Float16)v;
                    else if (MODE == 1) ((float*)Cout)[o] = v;
                    else                ((float*)Cout)[o] = resid[o] + gamma[n] * v;
                }
            }
        }
    }
}

// ------------------------ deformable-attn bilinear gather ------------------------
__global__ void msda_gather_kernel(const _Float16* __restrict__ value, // (N,LIN,768) f16
                                   const float* __restrict__ mix,      // (N*LQ, 256)
                                   const float* __restrict__ refp,     // (N,LQ,3,2)
                                   _Float16* __restrict__ attn)        // (N*LQ, 768) f16
{
    const int lane = threadIdx.x & 31;
    const int wid  = blockIdx.x * 8 + (threadIdx.x >> 5);
    const int h    = wid % HH;
    const int rq   = wid / HH;
    const int n    = rq >> 12;
    const float* mrow = mix + (size_t)rq * 256;

    float lg[12];
    float mx = -3.0e38f;
#pragma unroll
    for (int i = 0; i < 12; i++) { lg[i] = mrow[144 + h * 12 + i]; mx = fmaxf(mx, lg[i]); }
    float ssum = 0.f;
#pragma unroll
    for (int i = 0; i < 12; i++) { lg[i] = __expf(lg[i] - mx); ssum += lg[i]; }
    const float inv = 1.0f / ssum;

    float a0 = 0.f, a1 = 0.f, a2 = 0.f, a3 = 0.f;
    const int HS[3]  = {128, 64, 32};
    const int WSs[3] = {128, 64, 32};
    const int ST[3]  = {0, 16384, 20480};
    const int ch = h * DD + lane * 4;
#pragma unroll
    for (int l = 0; l < 3; l++) {
        const int Wi = WSs[l], Hi = HS[l];
        const float Wf = (float)Wi, Hf = (float)Hi;
        const float rx = refp[((size_t)rq * 3 + l) * 2 + 0];
        const float ry = refp[((size_t)rq * 3 + l) * 2 + 1];
        const size_t lvlbase = (size_t)n * LIN + ST[l];
#pragma unroll
        for (int p = 0; p < 4; p++) {
            const float ox = mrow[((h * 3 + l) * 4 + p) * 2 + 0];
            const float oy = mrow[((h * 3 + l) * 4 + p) * 2 + 1];
            const float x = (rx + ox / Wf) * Wf - 0.5f;
            const float y = (ry + oy / Hf) * Hf - 0.5f;
            const float x0f = floorf(x), y0f = floorf(y);
            const float wx = x - x0f, wy = y - y0f;
            const int x0 = (int)x0f, y0 = (int)y0f;
            const float aw = lg[l * 4 + p] * inv;
            const float w00 = (1.f - wx) * (1.f - wy) * aw;
            const float w10 = wx * (1.f - wy) * aw;
            const float w01 = (1.f - wx) * wy * aw;
            const float w11 = wx * wy * aw;
#pragma unroll
            for (int corner = 0; corner < 4; corner++) {
                const int xi = x0 + (corner & 1);
                const int yi = y0 + (corner >> 1);
                const float cw = (corner == 0) ? w00 : (corner == 1) ? w10
                                 : (corner == 2) ? w01 : w11;
                if (xi >= 0 && xi < Wi && yi >= 0 && yi < Hi) {
                    const size_t idx = (lvlbase + (size_t)yi * Wi + xi) * CC + ch;
                    v4h v = *(const v4h*)(value + idx);
                    a0 += cw * (float)v[0]; a1 += cw * (float)v[1];
                    a2 += cw * (float)v[2]; a3 += cw * (float)v[3];
                }
            }
        }
    }
    const size_t ob = (size_t)rq * CC + ch;
    v4h o; o[0] = (_Float16)a0; o[1] = (_Float16)a1; o[2] = (_Float16)a2; o[3] = (_Float16)a3;
    *(v4h*)(attn + ob) = o;
}

// ------------------------ host launcher ------------------------
extern "C" void kernel_launch(void* const* d_in, const int* in_sizes, int n_in,
                              void* d_out, int out_size, void* d_ws, size_t ws_size,
                              hipStream_t stream)
{
    (void)in_sizes; (void)n_in; (void)out_size; (void)ws_size;
    const float* query = (const float*)d_in[0];
    const float* refp  = (const float*)d_in[1];
    const float* feat  = (const float*)d_in[2];
    const float* qn_w  = (const float*)d_in[5];
    const float* qn_b  = (const float*)d_in[6];
    const float* fn_w  = (const float*)d_in[7];
    const float* fn_b  = (const float*)d_in[8];
    const float* so_w  = (const float*)d_in[9];
    const float* so_b  = (const float*)d_in[10];
    const float* aw_w  = (const float*)d_in[11];
    const float* aw_b  = (const float*)d_in[12];
    const float* vp_w  = (const float*)d_in[13];
    const float* vp_b  = (const float*)d_in[14];
    const float* op_w  = (const float*)d_in[15];
    const float* op_b  = (const float*)d_in[16];
    const float* gamma = (const float*)d_in[17];
    float* out = (float*)d_out;

    const int MQ = NBV * LQV;   // 16384
    const int MV = NBV * LIN;   // 86016
    const int NC = 256;         // padded combined projection width

    char* ws = (char*)d_ws;
    size_t off = 0;
    auto carve = [&](size_t bytes) -> void* {
        void* p = ws + off;
        off = (off + bytes + 255) & ~(size_t)255;
        return p;
    };
    _Float16* vp_h  = (_Float16*)carve((size_t)CC * CC * 2);
    _Float16* op_h  = (_Float16*)carve((size_t)CC * CC * 2);
    _Float16* wc_h  = (_Float16*)carve((size_t)NC * CC * 2);
    float*    cb    = (float*)   carve((size_t)NC * 4);
    _Float16* q_h   = (_Float16*)carve((size_t)MQ * CC * 2);
    _Float16* f_h   = (_Float16*)carve((size_t)MV * CC * 2);
    _Float16* val_h = (_Float16*)carve((size_t)MV * CC * 2);
    float*    mix   = (float*)   carve((size_t)MQ * NC * 4);
    _Float16* att_h = (_Float16*)carve((size_t)MQ * CC * 2);

    // allow ~97KB dynamic LDS for the GEMM kernels (host-side, deterministic)
    hipFuncSetAttribute(reinterpret_cast<const void*>(gemm_wmma_kernel<0>),
                        hipFuncAttributeMaxDynamicSharedMemorySize, GEMM_LDSB);
    hipFuncSetAttribute(reinterpret_cast<const void*>(gemm_wmma_kernel<1>),
                        hipFuncAttributeMaxDynamicSharedMemorySize, GEMM_LDSB);
    hipFuncSetAttribute(reinterpret_cast<const void*>(gemm_wmma_kernel<2>),
                        hipFuncAttributeMaxDynamicSharedMemorySize, GEMM_LDSB);

    // 1) LayerNorms -> f16 activations
    ln_f16_kernel<<<MQ, 256, 0, stream>>>(query, qn_w, qn_b, q_h);
    ln_f16_kernel<<<MV, 256, 0, stream>>>(feat,  fn_w, fn_b, f_h);

    // 2) weight conversions
    const int ncv = CC * CC;
    cvt_f16_kernel<<<(ncv + 255) / 256, 256, 0, stream>>>(vp_w, vp_h, ncv);
    cvt_f16_kernel<<<(ncv + 255) / 256, 256, 0, stream>>>(op_w, op_h, ncv);
    build_wc_kernel<<<(NC * CC + 255) / 256, 256, 0, stream>>>(so_w, aw_w, so_b, aw_b, wc_h, cb);

    // 3) value projection (86016x768x768) -> f16 value tensor
    gemm_wmma_kernel<0><<<MV / 64, 128, GEMM_LDSB, stream>>>(
        f_h, vp_h, vp_b, val_h, MV, CC, CC, nullptr, nullptr);

    // 4) fused offset/logit projection (16384x256x768) -> f32
    gemm_wmma_kernel<1><<<MQ / 64, 128, GEMM_LDSB, stream>>>(
        q_h, wc_h, cb, mix, MQ, NC, CC, nullptr, nullptr);

    // 5) softmax + bilinear gather core (one wave per (n,q,head))
    msda_gather_kernel<<<(MQ * HH) / 8, 256, 0, stream>>>(val_h, mix, refp, att_h);

    // 6) output projection + residual: out = query + gamma*(attn@op^T + op_b)
    gemm_wmma_kernel<2><<<MQ / 64, 128, GEMM_LDSB, stream>>>(
        att_h, op_h, op_b, out, MQ, CC, CC, query, gamma);
}
